// AttentionContextEncoder_75041668596419
// MI455X (gfx1250) — compile-verified, matrix operand order checked by reading, not run on hardware
//
#include <hip/hip_runtime.h>
#include <hip/hip_bf16.h>

// Problem constants (from reference)
#define B_    128
#define N_    16
#define F_    20
#define D_    4
#define E_    256     // nembed
#define H_    128     // half embed
#define NB_   10
#define ROWS_ (B_ * N_)        // 2048 GRU "batch" rows
#define G3_   (3 * E_)         // 768 gate width
#define SROW_ (E_ + 8)         // padded LDS row stride in halfs (528B -> bank rotate)

typedef __attribute__((ext_vector_type(16))) _Float16 v16h;
typedef __attribute__((ext_vector_type(8)))  float    v8f;
typedef __attribute__((ext_vector_type(4)))  unsigned int v4u;

union Frag16 { v16h h; v4u u[2]; };

// ---------------------------------------------------------------------------
// Kernel 1: pack w_ih / w_hh (f32, (256,768) row-major) into WMMA B-fragment
// layout: wpack[mat][nt(48)][kt(8)][lane(32)][16 halfs].
// B is 32x16 (KxN) per fragment; 16-bit layout (ISA 7.12.2):
//   lanes 0-15: N=lane, elems 0..7 -> K=0..7,  elems 8..15 -> K=16..23
//   lanes16-31: N=lane-16, elems 0..7 -> K=8..15, elems 8..15 -> K=24..31
// ---------------------------------------------------------------------------
__global__ __launch_bounds__(256) void pack_w(const float* __restrict__ w_ih,
                                              const float* __restrict__ w_hh,
                                              _Float16* __restrict__ wpack) {
    int idx = blockIdx.x * 256 + threadIdx.x;
    const int per_mat = 48 * 8 * 32 * 16;        // 196608 = 768*256
    if (idx >= 2 * per_mat) return;
    int mat  = idx / per_mat;
    int r    = idx % per_mat;
    int nt   = r >> 12;          // /4096
    int kt   = (r >> 9) & 7;
    int lane = (r >> 4) & 31;
    int h    = r & 15;
    int kg   = lane >> 4;
    int K    = (h < 8) ? (kg * 8 + h) : (16 + kg * 8 + (h - 8));
    int k    = kt * 32 + K;
    int n    = nt * 16 + (lane & 15);
    const float* w = mat ? w_hh : w_ih;
    wpack[idx] = (_Float16)w[k * G3_ + n];
}

// ---------------------------------------------------------------------------
// Kernel 2: fold biases.  bias[0..255]=b_r, [256..511]=b_z, [512..767]=b_in,
// [768..1023]=b_hn   (torch gate order r,z,n)
// ---------------------------------------------------------------------------
__global__ __launch_bounds__(256) void pack_bias(const float* __restrict__ b_ih,
                                                 const float* __restrict__ b_hh,
                                                 float* __restrict__ bias) {
    int e = threadIdx.x;
    bias[e]           = b_ih[e]         + b_hh[e];
    bias[E_ + e]      = b_ih[E_ + e]    + b_hh[E_ + e];
    bias[2 * E_ + e]  = b_ih[2 * E_ + e];
    bias[3 * E_ + e]  = b_hh[2 * E_ + e];
}

// ---------------------------------------------------------------------------
// Kernel 3: zero initial hidden state (both f16 and f32 copies)
// ---------------------------------------------------------------------------
__global__ __launch_bounds__(256) void zero_h(_Float16* __restrict__ h16,
                                              float* __restrict__ h32) {
    int i = blockIdx.x * 256 + threadIdx.x;
    if (i < ROWS_ * E_) { h16[i] = (_Float16)0.0f; h32[i] = 0.0f; }
}

// ---------------------------------------------------------------------------
// Kernel 4: build GRU inputs x[t][row][e] (f16), one block per (b,n,f).
//   e < 128 : prop  = tanh(ent . W_prop + b_prop)
//   e >=128 : rel   = sum_{j!=n} tanh([diff, dist] . W_rel + b_rel)
//   plus     bemb  = sum_k emb[bf[k]]
// ---------------------------------------------------------------------------
__global__ __launch_bounds__(256) void features(
    const float* __restrict__ entities,      // (B,N,F,D)
    const int*   __restrict__ bf,            // (B,N,F,NB)
    const float* __restrict__ W_prop,        // (D,H)
    const float* __restrict__ b_prop,        // (H)
    const float* __restrict__ W_rel,         // (D+1,H)
    const float* __restrict__ b_rel,         // (H)
    const float* __restrict__ emb,           // (2*NB, E)
    _Float16* __restrict__ xpack)            // (F, ROWS, E) f16
{
    int blk = blockIdx.x;                    // 0 .. B*N*F-1
    int b = blk / (N_ * F_);
    int rem = blk % (N_ * F_);
    int n = rem / F_;
    int f = rem % F_;
    int e = threadIdx.x;

    __shared__ float ent_s[N_][D_];
    __shared__ int   bf_s[NB_];

    if (e < N_ * D_) {
        int j = e >> 2, d = e & 3;
        ent_s[j][d] = entities[(((size_t)(b * N_ + j) * F_) + f) * D_ + d];
    }
    if (e < NB_) {
        bf_s[e] = bf[(((size_t)(b * N_ + n) * F_) + f) * NB_ + e];
    }
    __syncthreads();

    // embedding gather-sum
    float bacc = 0.0f;
    #pragma unroll
    for (int k = 0; k < NB_; ++k) bacc += emb[(size_t)bf_s[k] * E_ + e];

    float v;
    if (e < H_) {
        float t = b_prop[e];
        #pragma unroll
        for (int d = 0; d < D_; ++d) t += ent_s[n][d] * W_prop[d * H_ + e];
        v = tanhf(t);
    } else {
        int h = e - H_;
        float w0 = W_rel[0 * H_ + h], w1 = W_rel[1 * H_ + h];
        float w2 = W_rel[2 * H_ + h], w3 = W_rel[3 * H_ + h];
        float w4 = W_rel[4 * H_ + h], bb = b_rel[h];
        float acc = 0.0f;
        for (int j = 0; j < N_; ++j) {
            if (j == n) continue;
            float dx0 = ent_s[n][0] - ent_s[j][0];
            float dx1 = ent_s[n][1] - ent_s[j][1];
            float dx2 = ent_s[n][2] - ent_s[j][2];
            float dx3 = ent_s[n][3] - ent_s[j][3];
            float dist = sqrtf(dx0 * dx0 + dx1 * dx1);   // reference: only comps 0,1
            float t = bb + dx0 * w0 + dx1 * w1 + dx2 * w2 + dx3 * w3 + dist * w4;
            acc += tanhf(t);
        }
        v = acc;
    }

    size_t row = (size_t)b * N_ + n;
    xpack[((size_t)f * ROWS_ + row) * E_ + e] = (_Float16)(v + bacc);
}

// ---------------------------------------------------------------------------
// Fragment loaders
// ---------------------------------------------------------------------------
__device__ __forceinline__ v16h load_a_lds(const _Float16* row, int k0, int kg) {
    Frag16 f;
    f.u[0] = *(const v4u*)(row + k0 + kg * 8);        // K = k0 + kg*8 .. +7
    f.u[1] = *(const v4u*)(row + k0 + 16 + kg * 8);   // K = k0+16+kg*8 .. +7
    return f.h;
}
__device__ __forceinline__ v16h load_b_gl(const _Float16* wp, int nt, int kt, int lane) {
    const _Float16* p = wp + (((size_t)(nt * 8 + kt) * 32 + lane) << 4);
    Frag16 f;
    f.u[0] = *(const v4u*)(p);
    f.u[1] = *(const v4u*)(p + 8);
    return f.h;
}
#define WMMA_F16(A, Bm, C) \
    __builtin_amdgcn_wmma_f32_16x16x32_f16(false, (A), false, (Bm), (short)0, (C), false, false)

// ---------------------------------------------------------------------------
// Kernel 5: one GRU time step.  grid = (64 m-tiles, 4 e-tiles), 256 threads.
// Workgroup: 32 rows x 64 hidden cols.  Wave (wm, we) owns a 16x16 tile of
// each of the 4 gate accumulators:
//   acc_r  = x.Wih_r + h.Whh_r      acc_z = x.Wih_z + h.Whh_z
//   acc_in = x.Wih_n                acc_hn = h.Whh_n
// ---------------------------------------------------------------------------
__global__ __launch_bounds__(256, 2) void gru_step(
    const _Float16* __restrict__ xt,      // (ROWS, 256) f16
    const _Float16* __restrict__ wpack,   // 2 * (48,8,32,16) f16
    const float*    __restrict__ bias,    // 4*256
    const _Float16* __restrict__ hIn16,   // (ROWS,256) f16
    const float*    __restrict__ hIn32,   // (ROWS,256) f32
    _Float16* __restrict__ hOut16,
    float*    __restrict__ hOut32)
{
    // padded rows: 264 halfs = 528B = 132 banks -> each row rotates 4 banks,
    // so the 16 lanes of a ds_load_b128 A-fragment hit disjoint bank groups.
    __shared__ __align__(16) _Float16 sx[32 * SROW_];
    __shared__ __align__(16) _Float16 sh[32 * SROW_];

    const int tid  = threadIdx.x;
    const int lane = tid & 31;
    const int wid  = tid >> 5;
    const int wm   = wid & 1;     // m sub-tile (0/1)
    const int we   = wid >> 1;    // e sub-tile (0..3)
    const int m0   = blockIdx.x * 32;
    const int e0   = blockIdx.y * 64;

    const int ntbase = (e0 >> 4) + we;        // n-tile within 0..15
    const int ntr = ntbase;                   // r gate cols [0,256)
    const int ntz = 16 + ntbase;              // z gate cols [256,512)
    const int ntn = 32 + ntbase;              // n gate cols [512,768)
    const _Float16* wih = wpack;
    const _Float16* whh = wpack + (size_t)48 * 8 * 32 * 16;

    // Prefetch this wave's entire weight footprint (6 slices x 8KB) into the
    // near cache while LDS staging is in flight.  One instruction per slice:
    // lane stride 256B covers 8KB across 32 lanes.
    {
        const _Float16* pr[6] = {
            wih + (size_t)ntr * 4096, wih + (size_t)ntz * 4096, wih + (size_t)ntn * 4096,
            whh + (size_t)ntr * 4096, whh + (size_t)ntz * 4096, whh + (size_t)ntn * 4096 };
        #pragma unroll
        for (int s = 0; s < 6; ++s)
            __builtin_prefetch((const char*)pr[s] + (size_t)lane * 256, 0, 3);
    }

    // stage x_t and h tiles (32 rows x 256 halfs = 1024 uint4 each, padded rows)
    {
        const v4u* gx = (const v4u*)(xt    + (size_t)m0 * E_);
        const v4u* gh = (const v4u*)(hIn16 + (size_t)m0 * E_);
        v4u* lx = (v4u*)sx;
        v4u* lh = (v4u*)sh;
        #pragma unroll
        for (int i = 0; i < 4; ++i) {
            int idx = tid + i * 256;          // 0..1023
            int row = idx >> 5, col = idx & 31;
            int lidx = row * (SROW_ / 8) + col;   // 33 uint4 per padded row
            lx[lidx] = gx[idx];
            lh[lidx] = gh[idx];
        }
    }
    __syncthreads();

    const int kg   = lane >> 4;
    const int mrow = wm * 16 + (lane & 15);
    const _Float16* sxrow = sx + mrow * SROW_;
    const _Float16* shrow = sh + mrow * SROW_;

    v8f acc_r = {}, acc_z = {}, acc_in = {}, acc_hn = {};

    #pragma unroll
    for (int kt = 0; kt < 8; ++kt) {
        const int k0 = kt * 32;
        v16h ax = load_a_lds(sxrow, k0, kg);
        v16h ah = load_a_lds(shrow, k0, kg);
        v16h br_i = load_b_gl(wih, ntr, kt, lane);
        v16h bz_i = load_b_gl(wih, ntz, kt, lane);
        v16h bn_i = load_b_gl(wih, ntn, kt, lane);
        v16h br_h = load_b_gl(whh, ntr, kt, lane);
        v16h bz_h = load_b_gl(whh, ntz, kt, lane);
        v16h bn_h = load_b_gl(whh, ntn, kt, lane);
        acc_r  = WMMA_F16(ax, br_i, acc_r);
        acc_r  = WMMA_F16(ah, br_h, acc_r);
        acc_z  = WMMA_F16(ax, bz_i, acc_z);
        acc_z  = WMMA_F16(ah, bz_h, acc_z);
        acc_in = WMMA_F16(ax, bn_i, acc_in);
        acc_hn = WMMA_F16(ah, bn_h, acc_hn);
    }

    // Epilogue: gates.  D layout: col = lane&15; row = vr + 8*(lane>=16).
    const int e_abs  = e0 + we * 16 + (lane & 15);
    const int m_base = m0 + wm * 16 + 8 * (lane >> 4);
    const float br  = bias[e_abs];
    const float bz  = bias[E_ + e_abs];
    const float bin = bias[2 * E_ + e_abs];
    const float bhn = bias[3 * E_ + e_abs];

    #pragma unroll
    for (int vr = 0; vr < 8; ++vr) {
        const int m = m_base + vr;
        const size_t off = (size_t)m * E_ + e_abs;
        float r  = 1.0f / (1.0f + __expf(-(acc_r[vr] + br)));
        float z  = 1.0f / (1.0f + __expf(-(acc_z[vr] + bz)));
        float nn = tanhf(acc_in[vr] + bin + r * (acc_hn[vr] + bhn));
        float hold = hIn32[off];
        float hnew = (1.0f - z) * nn + z * hold;
        hOut32[off] = hnew;
        hOut16[off] = (_Float16)hnew;
    }
}

// ---------------------------------------------------------------------------
// Host side
// ---------------------------------------------------------------------------
extern "C" void kernel_launch(void* const* d_in, const int* in_sizes, int n_in,
                              void* d_out, int out_size, void* d_ws, size_t ws_size,
                              hipStream_t stream) {
    const float* entities = (const float*)d_in[0];
    const int*   bf       = (const int*)  d_in[1];
    const float* W_prop   = (const float*)d_in[2];
    const float* b_prop   = (const float*)d_in[3];
    const float* W_rel    = (const float*)d_in[4];
    const float* b_rel    = (const float*)d_in[5];
    const float* emb      = (const float*)d_in[6];
    const float* w_ih     = (const float*)d_in[7];
    const float* b_ih     = (const float*)d_in[8];
    const float* w_hh     = (const float*)d_in[9];
    const float* b_hh     = (const float*)d_in[10];
    (void)in_sizes; (void)n_in; (void)out_size; (void)ws_size;

    // workspace carve-up (256B aligned)
    char* ws = (char*)d_ws;
    size_t off = 0;
    auto carve = [&](size_t bytes) { char* p = ws + off; off = (off + bytes + 255) & ~(size_t)255; return p; };
    _Float16* xpack  = (_Float16*)carve((size_t)F_ * ROWS_ * E_ * 2);      // 21.0 MB
    _Float16* wpack  = (_Float16*)carve((size_t)2 * 48 * 8 * 32 * 16 * 2); // 0.79 MB
    float*    bias   = (float*)   carve((size_t)4 * E_ * 4);
    _Float16* h16[2] = { (_Float16*)carve((size_t)ROWS_ * E_ * 2),
                         (_Float16*)carve((size_t)ROWS_ * E_ * 2) };
    float*    h32[2] = { (float*)carve((size_t)ROWS_ * E_ * 4),
                         (float*)carve((size_t)ROWS_ * E_ * 4) };

    // 1) weight pack + bias fold + h0 = 0
    pack_w   <<<(2 * 48 * 8 * 32 * 16 + 255) / 256, 256, 0, stream>>>(w_ih, w_hh, wpack);
    pack_bias<<<1, 256, 0, stream>>>(b_ih, b_hh, bias);
    zero_h   <<<(ROWS_ * E_ + 255) / 256, 256, 0, stream>>>(h16[0], h32[0]);

    // 2) feature construction -> x (f16, time-major)
    features <<<B_ * N_ * F_, 256, 0, stream>>>(entities, bf, W_prop, b_prop,
                                                W_rel, b_rel, emb, xpack);

    // 3) 20 sequential GRU steps (ping-pong hidden state); last writes d_out
    for (int t = 0; t < F_; ++t) {
        const int cur = t & 1, nxt = cur ^ 1;
        const _Float16* xt = xpack + (size_t)t * ROWS_ * E_;
        float* outF32 = (t == F_ - 1) ? (float*)d_out : h32[nxt];
        gru_step<<<dim3(64, 4), 256, 0, stream>>>(xt, wpack, bias,
                                                  h16[cur], h32[cur],
                                                  h16[nxt], outF32);
    }
}
